// VelocityField_37177236914851
// MI455X (gfx1250) — compile-verified, matrix operand order checked by reading, not run on hardware
//
#include <hip/hip_runtime.h>

// ---------------------------------------------------------------------------
// Types for CDNA5 WMMA (gfx1250, wave32): D(f32 16x16) = A(bf16 16x32) x B(bf16 32x16) + C
// ---------------------------------------------------------------------------
typedef __bf16 bf16_t;
typedef __attribute__((ext_vector_type(16))) bf16_t v16bf;
typedef __attribute__((ext_vector_type(8)))  bf16_t v8bf;
typedef __attribute__((ext_vector_type(8)))  float  v8f;

#define WPB 2   // waves per block (64-thread blocks, 16-row tile per wave)

__device__ __forceinline__ float silu_f(float x) { return x / (1.0f + __expf(-x)); }

__device__ __forceinline__ v8f wmma_bf16(v16bf a, v16bf b, v8f c) {
  // 8 args: (neg_a, A, neg_b, B, c_mod, C, reuse_a, reuse_b)
  return __builtin_amdgcn_wmma_f32_16x16x32_bf16(false, a, false, b, (short)0, c,
                                                 false, false);
}

// A-operand lane layout for 16x16x32 bf16:
//   lane l: M = l&15 ; element e: K = (e>=8?16:0) + (l>=16?8:0) + (e&7)
// p points at rowBase + ktile*32 + (l>=16?8:0): elements 0..7 at p[0..7], 8..15 at p[16..23]
__device__ __forceinline__ v16bf load_a_g(const bf16_t* p) {
  v8bf lo = *(const v8bf*)(p);
  v8bf hi = *(const v8bf*)(p + 16);
  return __builtin_shufflevector(lo, hi, 0,1,2,3,4,5,6,7,8,9,10,11,12,13,14,15);
}

__device__ __forceinline__ v16bf load_a_f32(const float* p) {
  v16bf r;
  #pragma unroll
  for (int i = 0; i < 8; ++i) { r[i] = (bf16_t)p[i]; r[i + 8] = (bf16_t)p[16 + i]; }
  return r;
}

// B operand: pre-packed so that lane l's 16 bf16 values for tile t are contiguous.
__device__ __forceinline__ v16bf load_b_packed(const bf16_t* pk, int tile, int lane) {
  return *(const v16bf*)(pk + (((size_t)tile * 32 + lane) << 4));
}

// ---------------------------------------------------------------------------
// Weight packer: W (K x N, row-major f32) -> bf16 WMMA-B tiles.
// packed[((kt*NT+nt)*32 + lane)*16 + e] = W[kt*32 + kmap(lane,e)][nt*16 + (lane&15)]
// ---------------------------------------------------------------------------
__global__ void pack_w(const float* __restrict__ src, bf16_t* __restrict__ dst,
                       int K, int N) {
  int i = blockIdx.x * blockDim.x + threadIdx.x;
  if (i >= K * N) return;
  int e    = i & 15;
  int lane = (i >> 4) & 31;
  int t    = i >> 9;
  int NT   = N >> 4;
  int kt   = t / NT, nt = t % NT;
  int kk   = ((e >> 3) << 4) + ((lane >= 16) ? 8 : 0) + (e & 7);
  int k    = kt * 32 + kk;
  int n    = nt * 16 + (lane & 15);
  dst[i] = (bf16_t)src[(size_t)k * N + n];
}

__global__ void zero_kernel(float* p, int n) {
  int i = blockIdx.x * blockDim.x + threadIdx.x;
  if (i < n) p[i] = 0.f;
}

__global__ void vel_kernel(const float* __restrict__ xf, const float* __restrict__ pos,
                           float* __restrict__ out, int n) {
  int i = blockIdx.x * blockDim.x + threadIdx.x;
  if (i < n) out[i] = xf[i] - pos[i];
}

// ---------------------------------------------------------------------------
// Conditioning: h = [z[batch], silu(t*w1+b1)@w2+b2] @ cp_w + cp_b   (V x 128)
// ---------------------------------------------------------------------------
__global__ __launch_bounds__(128) void cond_kernel(
    const float* __restrict__ t, const float* __restrict__ z,
    const int* __restrict__ batch,
    const float* __restrict__ te_w1, const float* __restrict__ te_b1,
    const float* __restrict__ te_w2, const float* __restrict__ te_b2,
    const float* __restrict__ cp_w, const float* __restrict__ cp_b,
    float* __restrict__ h, bf16_t* __restrict__ hbf, int V) {
  const int vtx = blockIdx.x;
  const int f   = threadIdx.x;
  __shared__ float temb[16];
  __shared__ float zsh[64];
  const int b = batch[vtx];
  if (f < 16) {
    const float tt = t[b];
    float acc = te_b2[f];
    #pragma unroll
    for (int k = 0; k < 16; ++k) {
      float u = tt * te_w1[k] + te_b1[k];
      acc += silu_f(u) * te_w2[k * 16 + f];
    }
    temb[f] = acc;
  }
  if (f < 64) zsh[f] = z[b * 64 + f];
  __syncthreads();
  float acc = cp_b[f];
  #pragma unroll 8
  for (int k = 0; k < 64; ++k) acc += zsh[k] * cp_w[k * 128 + f];
  #pragma unroll
  for (int k = 0; k < 16; ++k) acc += temb[k] * cp_w[(64 + k) * 128 + f];
  const size_t idx = (size_t)vtx * 128 + f;
  h[idx]   = acc;
  hbf[idx] = (bf16_t)acc;
}

// ---------------------------------------------------------------------------
// Fused edge kernel: one 16-edge tile per wave.
//   m1  = silu([h_row,h_col] @ We1[0:256] + d2 * We1[256] + be1)   (WMMA, K=256)
//   m   = silu(m1 @ We2 + be2)                                     (WMMA, K=128)
//   coef= silu(m @ Wc1 + bc1) @ Wc2                                (WMMA + shfl reduce)
//   atomic: x_nxt[row] += rel*coef ; agg[row] += m
// ---------------------------------------------------------------------------
__global__ __launch_bounds__(32 * WPB) void egnn_edge_kernel(
    const int* __restrict__ eRow, const int* __restrict__ eCol,
    const float* __restrict__ x_cur, float* __restrict__ x_nxt,
    const bf16_t* __restrict__ hbf, float* __restrict__ agg,
    const bf16_t* __restrict__ We1p, const float* __restrict__ We1last,
    const float* __restrict__ be1,
    const bf16_t* __restrict__ We2p, const float* __restrict__ be2,
    const bf16_t* __restrict__ Wc1p, const float* __restrict__ bc1,
    const float* __restrict__ Wc2, int E) {
  __shared__ float m1s[WPB][16][128];
  __shared__ float ms [WPB][16][128];
  __shared__ int   rIdx[WPB][16];
  __shared__ float d2s [WPB][16];
  __shared__ float coefs[WPB][16];

  const int lane = threadIdx.x & 31;
  const int w    = threadIdx.x >> 5;
  const int tile = blockIdx.x * WPB + w;
  if (tile * 16 >= E) return;
  const int half = (lane >= 16) ? 8 : 0;
  const int ln   = lane & 15;

  const int e = tile * 16 + ln;
  const int r = eRow[e];
  const int c = eCol[e];

  float relx = 0.f, rely = 0.f, relz = 0.f;
  if (lane < 16) {
    relx = x_cur[r * 3 + 0] - x_cur[c * 3 + 0];
    rely = x_cur[r * 3 + 1] - x_cur[c * 3 + 1];
    relz = x_cur[r * 3 + 2] - x_cur[c * 3 + 2];
    d2s[w][ln]  = relx * relx + rely * rely + relz * relz;
    rIdx[w][ln] = r;
  }
  __builtin_amdgcn_wave_barrier();

  float d2v[8];
  #pragma unroll
  for (int v = 0; v < 8; ++v) d2v[v] = d2s[w][v + half];

  // ---- stage 1
  const bf16_t* hr = hbf + (size_t)r * 128;
  const bf16_t* hc = hbf + (size_t)c * 128;
  v16bf a1[8];
  #pragma unroll
  for (int kt = 0; kt < 4; ++kt) {
    a1[kt]     = load_a_g(hr + kt * 32 + half);
    a1[kt + 4] = load_a_g(hc + kt * 32 + half);
  }
  #pragma unroll
  for (int nt = 0; nt < 8; ++nt) {
    v8f acc = {};
    #pragma unroll
    for (int kt = 0; kt < 8; ++kt)
      acc = wmma_bf16(a1[kt], load_b_packed(We1p, kt * 8 + nt, lane), acc);
    const int n = nt * 16 + ln;
    const float b = be1[n], wl = We1last[n];
    #pragma unroll
    for (int v = 0; v < 8; ++v)
      m1s[w][v + half][n] = silu_f(acc[v] + b + d2v[v] * wl);
  }
  __builtin_amdgcn_wave_barrier();

  // ---- stage 2
  v16bf a2[4];
  #pragma unroll
  for (int kt = 0; kt < 4; ++kt)
    a2[kt] = load_a_f32(&m1s[w][ln][kt * 32 + half]);
  #pragma unroll
  for (int nt = 0; nt < 8; ++nt) {
    v8f acc = {};
    #pragma unroll
    for (int kt = 0; kt < 4; ++kt)
      acc = wmma_bf16(a2[kt], load_b_packed(We2p, kt * 8 + nt, lane), acc);
    const int n = nt * 16 + ln;
    const float b = be2[n];
    #pragma unroll
    for (int v = 0; v < 8; ++v)
      ms[w][v + half][n] = silu_f(acc[v] + b);
  }
  __builtin_amdgcn_wave_barrier();

  // ---- stage 3 + coef reduction
  v16bf a3[4];
  #pragma unroll
  for (int kt = 0; kt < 4; ++kt)
    a3[kt] = load_a_f32(&ms[w][ln][kt * 32 + half]);
  float cp[8] = {0.f, 0.f, 0.f, 0.f, 0.f, 0.f, 0.f, 0.f};
  #pragma unroll
  for (int nt = 0; nt < 8; ++nt) {
    v8f acc = {};
    #pragma unroll
    for (int kt = 0; kt < 4; ++kt)
      acc = wmma_bf16(a3[kt], load_b_packed(Wc1p, kt * 8 + nt, lane), acc);
    const int n = nt * 16 + ln;
    const float b = bc1[n], w2 = Wc2[n];
    #pragma unroll
    for (int v = 0; v < 8; ++v)
      cp[v] += silu_f(acc[v] + b) * w2;
  }
  #pragma unroll
  for (int off = 1; off < 16; off <<= 1) {
    #pragma unroll
    for (int v = 0; v < 8; ++v)
      cp[v] += __shfl_xor(cp[v], off, 32);
  }
  if (ln == 0) {
    #pragma unroll
    for (int v = 0; v < 8; ++v) coefs[w][v + half] = cp[v];
  }
  __builtin_amdgcn_wave_barrier();

  // ---- scatter x += rel * coef (segment_sum over row)
  if (lane < 16) {
    const float cf = coefs[w][ln];
    atomicAdd(&x_nxt[r * 3 + 0], relx * cf);
    atomicAdd(&x_nxt[r * 3 + 1], rely * cf);
    atomicAdd(&x_nxt[r * 3 + 2], relz * cf);
  }
  // ---- scatter agg += m
  #pragma unroll 4
  for (int i = 0; i < 64; ++i) {
    const int flat = lane + 32 * i;   // 16 edges x 128 feats = 2048
    const int ee   = flat >> 7;
    const int f    = flat & 127;
    atomicAdd(&agg[(size_t)rIdx[w][ee] * 128 + f], ms[w][ee][f]);
  }
}

// ---------------------------------------------------------------------------
// Node kernel: one 16-node tile per wave.
//   u = silu([h, agg] @ Wn1 + bn1);  h += u @ Wn2 + bn2   (in-place, tile-local)
// ---------------------------------------------------------------------------
__global__ __launch_bounds__(32 * WPB) void egnn_node_kernel(
    const float* __restrict__ agg, float* __restrict__ h, bf16_t* __restrict__ hbf,
    const bf16_t* __restrict__ Wn1p, const float* __restrict__ bn1,
    const bf16_t* __restrict__ Wn2p, const float* __restrict__ bn2, int V) {
  __shared__ float us[WPB][16][128];
  const int lane = threadIdx.x & 31;
  const int w    = threadIdx.x >> 5;
  const int tile = blockIdx.x * WPB + w;
  if (tile * 16 >= V) return;
  const int half = (lane >= 16) ? 8 : 0;
  const int ln   = lane & 15;
  const int node = tile * 16 + ln;

  v16bf a[8];
  const bf16_t* hr = hbf + (size_t)node * 128;
  const float*  ar = agg + (size_t)node * 128;
  #pragma unroll
  for (int kt = 0; kt < 4; ++kt) {
    a[kt]     = load_a_g(hr + kt * 32 + half);
    a[kt + 4] = load_a_f32(ar + kt * 32 + half);
  }
  #pragma unroll
  for (int nt = 0; nt < 8; ++nt) {
    v8f acc = {};
    #pragma unroll
    for (int kt = 0; kt < 8; ++kt)
      acc = wmma_bf16(a[kt], load_b_packed(Wn1p, kt * 8 + nt, lane), acc);
    const int n = nt * 16 + ln;
    const float b = bn1[n];
    #pragma unroll
    for (int v = 0; v < 8; ++v)
      us[w][v + half][n] = silu_f(acc[v] + b);
  }
  __builtin_amdgcn_wave_barrier();

  v16bf a2[4];
  #pragma unroll
  for (int kt = 0; kt < 4; ++kt)
    a2[kt] = load_a_f32(&us[w][ln][kt * 32 + half]);
  #pragma unroll
  for (int nt = 0; nt < 8; ++nt) {
    v8f acc = {};
    #pragma unroll
    for (int kt = 0; kt < 4; ++kt)
      acc = wmma_bf16(a2[kt], load_b_packed(Wn2p, kt * 8 + nt, lane), acc);
    const int n = nt * 16 + ln;
    const float b = bn2[n];
    #pragma unroll
    for (int v = 0; v < 8; ++v) {
      const size_t idx = (size_t)(tile * 16 + half + v) * 128 + n;
      const float hn = h[idx] + acc[v] + b;
      h[idx]   = hn;
      hbf[idx] = (bf16_t)hn;
    }
  }
}

// ---------------------------------------------------------------------------
// Host-side orchestration (graph-capture safe; all on `stream`)
// ---------------------------------------------------------------------------
extern "C" void kernel_launch(void* const* d_in, const int* in_sizes, int n_in,
                              void* d_out, int out_size, void* d_ws, size_t ws_size,
                              hipStream_t stream) {
  const float* pos   = (const float*)d_in[0];
  const float* t     = (const float*)d_in[1];
  const float* z     = (const float*)d_in[2];
  const int*   eidx  = (const int*)d_in[3];
  const int*   batch = (const int*)d_in[4];
  const float* te_w1 = (const float*)d_in[5];
  const float* te_b1 = (const float*)d_in[6];
  const float* te_w2 = (const float*)d_in[7];
  const float* te_b2 = (const float*)d_in[8];
  const float* cp_w  = (const float*)d_in[9];
  const float* cp_b  = (const float*)d_in[10];
  const float* We1   = (const float*)d_in[11];
  const float* be1   = (const float*)d_in[12];
  const float* We2   = (const float*)d_in[13];
  const float* be2   = (const float*)d_in[14];
  const float* Wc1   = (const float*)d_in[15];
  const float* bc1   = (const float*)d_in[16];
  const float* Wc2   = (const float*)d_in[17];
  const float* Wn1   = (const float*)d_in[18];
  const float* bn1   = (const float*)d_in[19];
  const float* Wn2   = (const float*)d_in[20];
  const float* bn2   = (const float*)d_in[21];

  const int V  = in_sizes[4];
  const int E  = in_sizes[3] / 2;
  const int NL = 4;
  const int* eRow = eidx;
  const int* eCol = eidx + E;

  // ---- workspace layout (~54 MB) ----
  char* ws = (char*)d_ws;
  size_t off = 0;
  auto salloc = [&](size_t bytes) -> char* {
    char* p = ws + off;
    off += (bytes + 255) & ~(size_t)255;
    return p;
  };
  float*  h    = (float*)salloc((size_t)V * 128 * sizeof(float));
  bf16_t* hbf  = (bf16_t*)salloc((size_t)V * 128 * sizeof(bf16_t));
  float*  agg  = (float*)salloc((size_t)V * 128 * sizeof(float));
  float*  xA   = (float*)salloc((size_t)V * 3 * sizeof(float));
  float*  xB   = (float*)salloc((size_t)V * 3 * sizeof(float));
  const size_t PL = 114688;  // packed bf16 elements per layer
  bf16_t* pack = (bf16_t*)salloc((size_t)NL * PL * sizeof(bf16_t));
  (void)ws_size; (void)n_in; (void)out_size;

  // ---- pack all layer weights into bf16 WMMA-B tiles ----
  for (int l = 0; l < NL; ++l) {
    bf16_t* pl   = pack + (size_t)l * PL;
    bf16_t* we1p = pl;
    bf16_t* we2p = we1p + 32768;
    bf16_t* wc1p = we2p + 16384;
    bf16_t* wn1p = wc1p + 16384;
    bf16_t* wn2p = wn1p + 32768;
    pack_w<<<(256 * 128 + 255) / 256, 256, 0, stream>>>(We1 + (size_t)l * 257 * 128, we1p, 256, 128);
    pack_w<<<(128 * 128 + 255) / 256, 256, 0, stream>>>(We2 + (size_t)l * 128 * 128, we2p, 128, 128);
    pack_w<<<(128 * 128 + 255) / 256, 256, 0, stream>>>(Wc1 + (size_t)l * 128 * 128, wc1p, 128, 128);
    pack_w<<<(256 * 128 + 255) / 256, 256, 0, stream>>>(Wn1 + (size_t)l * 256 * 128, wn1p, 256, 128);
    pack_w<<<(128 * 128 + 255) / 256, 256, 0, stream>>>(Wn2 + (size_t)l * 128 * 128, wn2p, 128, 128);
  }

  // ---- conditioning ----
  cond_kernel<<<V, 128, 0, stream>>>(t, z, batch, te_w1, te_b1, te_w2, te_b2,
                                     cp_w, cp_b, h, hbf, V);

  // ---- layers ----
  const float* cur = pos;
  float* xbuf[2] = {xA, xB};
  const int eTiles = (E + 15) / 16;
  const int vTiles = (V + 15) / 16;
  for (int l = 0; l < NL; ++l) {
    float* nxt = xbuf[l & 1];
    hipMemcpyAsync(nxt, cur, (size_t)V * 3 * sizeof(float),
                   hipMemcpyDeviceToDevice, stream);
    zero_kernel<<<(V * 128 + 255) / 256, 256, 0, stream>>>(agg, V * 128);

    bf16_t* pl   = pack + (size_t)l * PL;
    bf16_t* we1p = pl;
    bf16_t* we2p = we1p + 32768;
    bf16_t* wc1p = we2p + 16384;
    bf16_t* wn1p = wc1p + 16384;
    bf16_t* wn2p = wn1p + 32768;

    egnn_edge_kernel<<<(eTiles + WPB - 1) / WPB, 32 * WPB, 0, stream>>>(
        eRow, eCol, cur, nxt, hbf, agg,
        we1p, We1 + (size_t)l * 257 * 128 + (size_t)256 * 128, be1 + l * 128,
        we2p, be2 + l * 128, wc1p, bc1 + l * 128, Wc2 + l * 128, E);

    egnn_node_kernel<<<(vTiles + WPB - 1) / WPB, 32 * WPB, 0, stream>>>(
        agg, h, hbf, wn1p, bn1 + l * 128, wn2p, bn2 + l * 128, V);

    cur = nxt;
  }

  vel_kernel<<<(V * 3 + 255) / 256, 256, 0, stream>>>(cur, pos, (float*)d_out, V * 3);
}
